// PEER_15212774163100
// MI455X (gfx1250) — matrix operand order repaired and checked by reading, not compiled, not run on hardware
//
#include <hip/hip_runtime.h>
#include <hip/hip_bf16.h>

// ---------------------------------------------------------------------------
// PEER layer for MI455X (gfx1250, wave32).
// Pipeline: bf16 WMMA projection GEMM (32x32 per wave, 2x2 WMMA tiles) ->
// BN stats -> BN+L2-normalize -> bf16 WMMA sub-key scoring -> exact product
// top-16 + softmax -> expert gather (async-to-LDS staging, dot / erf-GELU /
// weighted axpy) with LDS accumulation.
// ---------------------------------------------------------------------------

typedef __attribute__((ext_vector_type(16))) __bf16 v16bf;
typedef __attribute__((ext_vector_type(8)))  float  v8f;

static __device__ __forceinline__ unsigned short f32_to_bf16(float f) {
  unsigned int u = __float_as_uint(f);
  u += 0x7FFFu + ((u >> 16) & 1u);   // round-to-nearest-even
  return (unsigned short)(u >> 16);
}

// ---------------- conversion kernels ----------------

__global__ __launch_bounds__(256) void cvt_bf16_kernel(const float* __restrict__ in,
                                                       unsigned short* __restrict__ out,
                                                       int n) {
  int i = blockIdx.x * 256 + threadIdx.x;
  if (i < n) out[i] = f32_to_bf16(in[i]);
}

// Wq [1024 x 2048] f32 -> WqT [2048 x 1024] bf16 (B^T layout for WMMA B loads)
__global__ __launch_bounds__(256) void cvt_wqT_kernel(const float* __restrict__ Wq,
                                                      unsigned short* __restrict__ WqT) {
  int i = blockIdx.x * 256 + threadIdx.x;       // 0 .. 2M-1, i = d*2048 + n
  int d = i >> 11;
  int n = i & 2047;
  WqT[(size_t)n * 1024 + d] = f32_to_bf16(Wq[i]);
}

// ---------------- WMMA GEMM: C[M,N] = A[M,K] * Bt[N,K]^T (+bias) -------------
// A row-major bf16, Bt row-major bf16 (B stored transposed), C f32.
// One wave computes a 32x32 tile (2x2 WMMA tiles): each A fragment is reused
// across 2 column-tiles and each B fragment across 2 row-tiles.
// Block = 8 waves arranged 2(M) x 4(N) -> 64 x 128 block tile.
// Requires M%64==0, N%128==0, K%32==0.

__global__ __launch_bounds__(256) void wmma_gemm_bf16(const unsigned short* __restrict__ A,
                                                      const unsigned short* __restrict__ Bt,
                                                      const float* __restrict__ bias,
                                                      float* __restrict__ C,
                                                      int M, int N, int K) {
  const int lane  = threadIdx.x & 31;
  const int wave  = threadIdx.x >> 5;
  const int waveM = wave >> 2;                   // 0..1
  const int waveN = wave & 3;                    // 0..3
  const int m0 = blockIdx.y * 64 + waveM * 32;
  const int n0 = blockIdx.x * 128 + waveN * 32;

  // A fragment addressing (16-bit A 16x32): lane half selects K sub-offset 0/8
  const int aKoff = (lane >= 16) ? 8 : 0;
  // B fragment addressing: lane half selects K sub-offset 0/8 (same as A)
  const int bKoff = (lane >= 16) ? 8 : 0;

  const unsigned short* __restrict__ Arow0 = A + (size_t)(m0 + (lane & 15)) * K;
  const unsigned short* __restrict__ Arow1 = Arow0 + (size_t)16 * K;
  const unsigned short* __restrict__ Brow0 = Bt + (size_t)(n0 + (lane & 15)) * K;
  const unsigned short* __restrict__ Brow1 = Brow0 + (size_t)16 * K;

  v8f acc00 = {}, acc01 = {}, acc10 = {}, acc11 = {};
  for (int k0 = 0; k0 < K; k0 += 32) {
    union Frag { v16bf v; uint4 q[2]; } a0, a1, b0, b1;
    a0.q[0] = *(const uint4*)(Arow0 + k0 + aKoff);
    a0.q[1] = *(const uint4*)(Arow0 + k0 + 16 + aKoff);
    a1.q[0] = *(const uint4*)(Arow1 + k0 + aKoff);
    a1.q[1] = *(const uint4*)(Arow1 + k0 + 16 + aKoff);
    b0.q[0] = *(const uint4*)(Brow0 + k0 + bKoff);
    b0.q[1] = *(const uint4*)(Brow0 + k0 + 16 + bKoff);
    b1.q[0] = *(const uint4*)(Brow1 + k0 + bKoff);
    b1.q[1] = *(const uint4*)(Brow1 + k0 + 16 + bKoff);
    acc00 = __builtin_amdgcn_wmma_f32_16x16x32_bf16(false, a0.v, false, b0.v, (short)0, acc00, false, false);
    acc01 = __builtin_amdgcn_wmma_f32_16x16x32_bf16(false, a0.v, false, b1.v, (short)0, acc01, false, false);
    acc10 = __builtin_amdgcn_wmma_f32_16x16x32_bf16(false, a1.v, false, b0.v, (short)0, acc10, false, false);
    acc11 = __builtin_amdgcn_wmma_f32_16x16x32_bf16(false, a1.v, false, b1.v, (short)0, acc11, false, false);
  }

  // C/D layout: VGPR g -> row (g + lane>=16 ? 8 : 0); lane&15 -> col
  const int mBase = (lane >= 16) ? 8 : 0;
  const int col0 = n0 + (lane & 15);
  const int col1 = col0 + 16;
  const float bv0 = bias ? bias[col0] : 0.0f;
  const float bv1 = bias ? bias[col1] : 0.0f;
#pragma unroll
  for (int g = 0; g < 8; ++g) {
    const size_t r0 = (size_t)(m0 + mBase + g) * N;
    const size_t r1 = (size_t)(m0 + 16 + mBase + g) * N;
    C[r0 + col0] = acc00[g] + bv0;
    C[r0 + col1] = acc01[g] + bv1;
    C[r1 + col0] = acc10[g] + bv0;
    C[r1 + col1] = acc11[g] + bv1;
  }
}

// ---------------- BatchNorm batch statistics (training mode, biased var) -----

__global__ __launch_bounds__(256) void bn_stats_kernel(const float* __restrict__ q,
                                                       float* __restrict__ mean,
                                                       float* __restrict__ rstd) {
  const int c = blockIdx.x;        // channel 0..255
  const int tid = threadIdx.x;     // 256 threads
  float s = 0.f, ss = 0.f;
  for (int i = tid; i < 8192; i += 256) {             // samples = token*8 + h
    float x = q[(size_t)(i >> 3) * 2048 + (i & 7) * 256 + c];
    s += x; ss += x * x;
  }
  __shared__ float sh[256], sh2[256];
  sh[tid] = s; sh2[tid] = ss;
  __syncthreads();
  for (int off = 128; off > 0; off >>= 1) {
    if (tid < off) { sh[tid] += sh[tid + off]; sh2[tid] += sh2[tid + off]; }
    __syncthreads();
  }
  if (tid == 0) {
    float m = sh[0] * (1.0f / 8192.0f);
    float var = sh2[0] * (1.0f / 8192.0f) - m * m;
    mean[c] = m;
    rstd[c] = rsqrtf(var + 1e-5f);
  }
}

// ---------------- BN apply + split halves + L2 normalize -> bf16 -------------

__global__ __launch_bounds__(128) void q_norm_kernel(const float* __restrict__ q,
                                                     const float* __restrict__ mean,
                                                     const float* __restrict__ rstd,
                                                     const float* __restrict__ gamma,
                                                     const float* __restrict__ beta,
                                                     unsigned short* __restrict__ qn0,
                                                     unsigned short* __restrict__ qn1) {
  const int rh = blockIdx.x;          // token*8 + h, 0..8191
  const int tid = threadIdx.x;        // 128
  const int token = rh >> 3, h = rh & 7;
  const float* qp = q + (size_t)token * 2048 + h * 256;
  const int c0 = tid, c1 = tid + 128;
  float x0 = (qp[c0] - mean[c0]) * rstd[c0] * gamma[c0] + beta[c0];
  float x1 = (qp[c1] - mean[c1]) * rstd[c1] * gamma[c1] + beta[c1];
  __shared__ float sh0[128], sh1[128];
  sh0[tid] = x0 * x0; sh1[tid] = x1 * x1;
  __syncthreads();
  for (int off = 64; off > 0; off >>= 1) {
    if (tid < off) { sh0[tid] += sh0[tid + off]; sh1[tid] += sh1[tid + off]; }
    __syncthreads();
  }
  float s0 = 1.0f / fmaxf(sqrtf(sh0[0]), 1e-12f);
  float s1 = 1.0f / fmaxf(sqrtf(sh1[0]), 1e-12f);
  qn0[(size_t)rh * 128 + tid] = f32_to_bf16(x0 * s0);
  qn1[(size_t)rh * 128 + tid] = f32_to_bf16(x1 * s1);
}

// ---------------- sub-key row L2 normalize -> bf16 (already B^T layout) ------

__global__ __launch_bounds__(128) void key_norm_kernel(const float* __restrict__ keys,
                                                       unsigned short* __restrict__ kn) {
  const int r = blockIdx.x, tid = threadIdx.x;   // 128 x 128
  float x = keys[r * 128 + tid];
  __shared__ float sh[128];
  sh[tid] = x * x;
  __syncthreads();
  for (int off = 64; off > 0; off >>= 1) {
    if (tid < off) sh[tid] += sh[tid + off];
    __syncthreads();
  }
  float s = 1.0f / fmaxf(sqrtf(sh[0]), 1e-12f);
  kn[r * 128 + tid] = f32_to_bf16(x * s);
}

// ---------------- product top-16 + softmax (one wave per row) ----------------
// Exact: global top-16 of s0[i]+s1[j] requires i in top16(s0), j in top16(s1).

__global__ __launch_bounds__(32) void topk_softmax_kernel(const float* __restrict__ s0,
                                                          const float* __restrict__ s1,
                                                          float* __restrict__ wOut,
                                                          int* __restrict__ idxOut) {
  const int rh = blockIdx.x;
  const int lane = threadIdx.x;   // 32 = one wave32
  float v0[4], v1[4];
#pragma unroll
  for (int j = 0; j < 4; ++j) {
    v0[j] = s0[(size_t)rh * 128 + j * 32 + lane];
    v1[j] = s1[(size_t)rh * 128 + j * 32 + lane];
  }
  const float NEG = -3.4e38f;
  float t0v[16], t1v[16]; int t0i[16], t1i[16];

  for (int t = 0; t < 16; ++t) {          // top-16 of s0 half
    float bv = NEG; int bi = 0;
#pragma unroll
    for (int j = 0; j < 4; ++j) { if (v0[j] > bv) { bv = v0[j]; bi = j * 32 + lane; } }
    for (int off = 16; off > 0; off >>= 1) {
      float ov = __shfl_xor(bv, off, 32); int oi = __shfl_xor(bi, off, 32);
      if (ov > bv || (ov == bv && oi < bi)) { bv = ov; bi = oi; }
    }
    t0v[t] = bv; t0i[t] = bi;
    if ((bi & 31) == lane) v0[bi >> 5] = NEG;
  }
  for (int t = 0; t < 16; ++t) {          // top-16 of s1 half
    float bv = NEG; int bi = 0;
#pragma unroll
    for (int j = 0; j < 4; ++j) { if (v1[j] > bv) { bv = v1[j]; bi = j * 32 + lane; } }
    for (int off = 16; off > 0; off >>= 1) {
      float ov = __shfl_xor(bv, off, 32); int oi = __shfl_xor(bi, off, 32);
      if (ov > bv || (ov == bv && oi < bi)) { bv = ov; bi = oi; }
    }
    t1v[t] = bv; t1i[t] = bi;
    if ((bi & 31) == lane) v1[bi >> 5] = NEG;
  }

  // 256 candidate sums, 8 per lane; top-16 of candidates
  float c[8];
#pragma unroll
  for (int u = 0; u < 8; ++u) {
    int ci = u * 32 + lane;
    c[u] = t0v[ci >> 4] + t1v[ci & 15];
  }
  float tv[16]; int tc[16];
  for (int t = 0; t < 16; ++t) {
    float bv = NEG; int bi = 0;
#pragma unroll
    for (int u = 0; u < 8; ++u) { int ci = u * 32 + lane; if (c[u] > bv) { bv = c[u]; bi = ci; } }
    for (int off = 16; off > 0; off >>= 1) {
      float ov = __shfl_xor(bv, off, 32); int oi = __shfl_xor(bi, off, 32);
      if (ov > bv || (ov == bv && oi < bi)) { bv = ov; bi = oi; }
    }
    tv[t] = bv; tc[t] = bi;
    if ((bi & 31) == lane) c[bi >> 5] = NEG;
  }

  // softmax over the 16 winners (tv[0] is the max)
  float mx = tv[0], sum = 0.f, w[16];
#pragma unroll
  for (int t = 0; t < 16; ++t) { w[t] = __expf(tv[t] - mx); sum += w[t]; }
  float inv = 1.0f / sum;
  if (lane < 16) {
    int ci = tc[lane];
    wOut[(size_t)rh * 16 + lane]  = w[lane] * inv;
    idxOut[(size_t)rh * 16 + lane] = t0i[ci >> 4] * 128 + t1i[ci & 15];
  }
}

// ---------------- expert gather / GELU / weighted accumulate -----------------
// One block per token (256 thr = 8 waves); wave h handles its 16 experts.
// hidden vector staged to LDS with gfx1250 async-to-LDS DMA (ASYNCcnt path);
// expert tables (128 MB) are L2-resident on MI455X (192 MB L2).

__global__ __launch_bounds__(256) void expert_kernel(const float* __restrict__ hs,
                                                     const float* __restrict__ down,
                                                     const float* __restrict__ up,
                                                     const float* __restrict__ wIn,
                                                     const int* __restrict__ idxIn,
                                                     float* __restrict__ out) {
  __shared__ float hsm[1024];
  __shared__ float acc[1024];
  const int token = blockIdx.x;
  const int tid = threadIdx.x;

  // Async DMA the 4KB hidden vector into LDS: 256 lanes x 16B, one instruction.
  {
    unsigned lds_off = (unsigned)(size_t)(&hsm[0]) + (unsigned)tid * 16u;  // LDS byte addr
    const float* gsrc = hs + (size_t)token * 1024 + tid * 4;
    asm volatile("global_load_async_to_lds_b128 %0, %1, off"
                 :: "v"(lds_off), "v"(gsrc) : "memory");
  }
  for (int i = tid; i < 1024; i += 256) acc[i] = 0.f;
  asm volatile("s_wait_asynccnt 0" ::: "memory");
  __syncthreads();

  const int h = tid >> 5;       // wave id = head
  const int lane = tid & 31;
  float upacc[32];
#pragma unroll
  for (int i = 0; i < 32; ++i) upacc[i] = 0.f;

  for (int k = 0; k < 16; ++k) {
    const int slot = (token * 8 + h) * 16 + k;
    const int idx = idxIn[slot];
    const float wgt = wIn[slot];
    const float* __restrict__ dr = down + (size_t)idx * 1024;
    const float* __restrict__ ur = up + (size_t)idx * 1024;
    __builtin_prefetch(ur + lane, 0, 0);           // global_prefetch_b8 the up row

    float partial = 0.f;
#pragma unroll
    for (int i = 0; i < 32; ++i) partial += dr[lane + 32 * i] * hsm[lane + 32 * i];
    for (int off = 16; off > 0; off >>= 1) partial += __shfl_xor(partial, off, 32);

    float x = partial;
    float eo = 0.5f * x * (1.0f + erff(x * 0.70710678118654752f));  // exact GELU
    float scale = eo * wgt;
#pragma unroll
    for (int i = 0; i < 32; ++i) upacc[i] += ur[lane + 32 * i] * scale;
  }
#pragma unroll
  for (int i = 0; i < 32; ++i) atomicAdd(&acc[lane + 32 * i], upacc[i]);  // ds_add_f32
  __syncthreads();
  for (int i = tid; i < 1024; i += 256) out[(size_t)token * 1024 + i] = acc[i];
}

// ---------------------------------------------------------------------------

extern "C" void kernel_launch(void* const* d_in, const int* in_sizes, int n_in,
                              void* d_out, int out_size, void* d_ws, size_t ws_size,
                              hipStream_t stream) {
  const float* hs       = (const float*)d_in[0];  // [2,512,1024]
  const float* Wq       = (const float*)d_in[1];  // [1024,2048]
  const float* bq       = (const float*)d_in[2];  // [2048]
  const float* bn_gamma = (const float*)d_in[3];  // [256]
  const float* bn_beta  = (const float*)d_in[4];  // [256]
  const float* sk0      = (const float*)d_in[5];  // [128,128]
  const float* sk1      = (const float*)d_in[6];  // [128,128]
  const float* e_down   = (const float*)d_in[7];  // [16384,1024]
  const float* e_up     = (const float*)d_in[8];  // [16384,1024]
  float* out            = (float*)d_out;          // [2,512,1024]

  char* base = (char*)d_ws;
  size_t off = 0;
  auto alloc = [&](size_t bytes) -> char* {
    char* p = base + off;
    off = (off + bytes + 255) & ~(size_t)255;
    return p;
  };
  unsigned short* hs_bf  = (unsigned short*)alloc(1024u * 1024u * 2u);
  unsigned short* WqT_bf = (unsigned short*)alloc(2048u * 1024u * 2u);
  float*          qf     = (float*)alloc((size_t)1024 * 2048 * 4);
  float*          mean   = (float*)alloc(256 * 4);
  float*          rstd   = (float*)alloc(256 * 4);
  unsigned short* k0n    = (unsigned short*)alloc(128u * 128u * 2u);
  unsigned short* k1n    = (unsigned short*)alloc(128u * 128u * 2u);
  unsigned short* qn0    = (unsigned short*)alloc((size_t)8192 * 128 * 2);
  unsigned short* qn1    = (unsigned short*)alloc((size_t)8192 * 128 * 2);
  float*          s0     = (float*)alloc((size_t)8192 * 128 * 4);
  float*          s1     = (float*)alloc((size_t)8192 * 128 * 4);
  float*          wSel   = (float*)alloc((size_t)8192 * 16 * 4);
  int*            iSel   = (int*)alloc((size_t)8192 * 16 * 4);
  (void)ws_size; (void)in_sizes; (void)n_in; (void)out_size;

  // 1) convert hs -> bf16; Wq -> transposed bf16
  cvt_bf16_kernel<<<4096, 256, 0, stream>>>(hs, hs_bf, 1024 * 1024);
  cvt_wqT_kernel<<<8192, 256, 0, stream>>>(Wq, WqT_bf);

  // 2) projection GEMM: q[1024,2048] = hs[1024,1024] * Wq + bq  (WMMA bf16)
  wmma_gemm_bf16<<<dim3(2048 / 128, 1024 / 64), 256, 0, stream>>>(
      hs_bf, WqT_bf, bq, qf, 1024, 2048, 1024);

  // 3) BatchNorm batch statistics (8192 samples x 256 channels)
  bn_stats_kernel<<<256, 256, 0, stream>>>(qf, mean, rstd);

  // 4) BN apply + L2 normalize halves -> bf16
  q_norm_kernel<<<8192, 128, 0, stream>>>(qf, mean, rstd, bn_gamma, bn_beta, qn0, qn1);

  // 5) normalize sub-keys -> bf16 (row-major = B^T layout)
  key_norm_kernel<<<128, 128, 0, stream>>>(sk0, k0n);
  key_norm_kernel<<<128, 128, 0, stream>>>(sk1, k1n);

  // 6) scoring GEMMs: s = qn[8192,128] * kn^T (WMMA bf16)
  wmma_gemm_bf16<<<dim3(1, 8192 / 64), 256, 0, stream>>>(qn0, k0n, nullptr, s0, 8192, 128, 128);
  wmma_gemm_bf16<<<dim3(1, 8192 / 64), 256, 0, stream>>>(qn1, k1n, nullptr, s1, 8192, 128, 128);

  // 7) exact product top-16 + softmax (one wave per (token,head))
  topk_softmax_kernel<<<8192, 32, 0, stream>>>(s0, s1, wSel, iSel);

  // 8) expert gather / GELU / weighted accumulation
  expert_kernel<<<1024, 256, 0, stream>>>(hs, e_down, e_up, wSel, iSel, out);
}
